// SpikingV2Context_19155554140646
// MI455X (gfx1250) — matrix operand order repaired and checked by reading, not compiled, not run on hardware
//
#include <hip/hip_runtime.h>
#include <stdint.h>

// ---------------------------------------------------------------------------
// MI455X (gfx1250) fused spiking-V2 step.
//   GEMM1: [4096 x 4640]bf16 x [4640 x 4096]bf16 -> fp32, fused spike update
//   GEMM2: [4096 x 4096]bf16 x [4096 x 384]bf16 -> fp32 (+bias)
//   Finalize: softmax / softplus epilogues.
// bf16 WMMA (v_wmma_f32_16x16x32_bf16) w/ fp32 accumulation. Tiles staged to
// LDS via GLOBAL_LOAD_ASYNC_TO_LDS_B128 (ASYNCcnt) with double buffering:
// one barrier per K-step, loads fully overlapped with the WMMA stream.
// ---------------------------------------------------------------------------

typedef __attribute__((ext_vector_type(16))) __bf16 v16bf;
typedef __attribute__((ext_vector_type(8)))  __bf16 bf16x8;
typedef __attribute__((ext_vector_type(8)))  float  v8f;

#define B_BATCH 4096
#define NN      180
#define NV2     4096
#define IN_DIM  542
#define K1PAD   544            // IN_DIM padded to multiple of 32
#define KTOT    4640           // K1PAD + NV2 = 145 * 32
#define KSTEPS1 (KTOT / 32)
#define N2PAD   384            // 180 + 180 + 1 padded to multiple of 128
#define KSTEPS2 (NV2 / 32)

#define BETA_MEM   0.95122942450071400910f   // exp(-1/20)
#define RHO_AD     0.99501247919268231420f   // exp(-1/200)
#define ONE_M_RHO  0.00498752080731768580f
#define BETA_AD    1.8f
#define ALPHA_F    0.9f
#define PI_MAX     10.0f

// d_out flat offsets (floats), tuple order: mu, pi, fb, v, z, x, b
#define OFF_MU ((size_t)0)
#define OFF_PI ((size_t)737280)
#define OFF_FB ((size_t)741376)
#define OFF_V  ((size_t)1478656)
#define OFF_Z  ((size_t)18255872)
#define OFF_X  ((size_t)35033088)
#define OFF_B  ((size_t)51810304)

// LDS tile: 128 rows x 32 bf16, row stride padded to 40 halfwords (80 B:
// 16 B aligned for b128 fragment reads, 20-bank stagger between rows).
#define LDS_STRIDE 40

static __device__ __forceinline__ uint16_t f2bf(float f) {
    union { float f; uint32_t u; } c; c.f = f;
    uint32_t u = c.u;
    uint32_t r = u + 0x7FFFu + ((u >> 16) & 1u);   // round-to-nearest-even
    return (uint16_t)(r >> 16);
}

// CDNA5 async direct global->LDS copy, 16 B per lane, tracked by ASYNCcnt.
static __device__ __forceinline__ void async_ld_b128(uint32_t lds_addr,
                                                     uint64_t gaddr) {
    asm volatile("global_load_async_to_lds_b128 %0, %1, off"
                 :: "v"(lds_addr), "v"(gaddr) : "memory");
}
static __device__ __forceinline__ void wait_async0() {
    asm volatile("s_wait_asynccnt 0" ::: "memory");
}
// Low 32 bits of a generic shared pointer = LDS byte offset (ISA 10.2).
static __device__ __forceinline__ uint32_t lds_addr_of(const void* p) {
    return (uint32_t)(uintptr_t)p;
}

// ---------------------------------------------------------------------------
// Pack kernels: fp32 -> bf16, fuse the 4-way input concat + zero padding.
// ---------------------------------------------------------------------------
__global__ void pack_A_kernel(const float* __restrict__ x_l4,
                              const float* __restrict__ x_l23,
                              const float* __restrict__ cue,
                              const float* __restrict__ task_state,
                              const float* __restrict__ x_prev,
                              uint16_t* __restrict__ Apack) {
    size_t idx = (size_t)blockIdx.x * blockDim.x + threadIdx.x;
    if (idx >= (size_t)B_BATCH * KTOT) return;
    int row = (int)(idx / KTOT);
    int col = (int)(idx % KTOT);
    float val;
    if (col < 180)            val = x_l4[(size_t)row * NN + col];
    else if (col < 360)       val = x_l23[(size_t)row * NN + (col - 180)];
    else if (col < 540)       val = cue[(size_t)row * NN + (col - 360)];
    else if (col < 542)       val = task_state[(size_t)row * 2 + (col - 540)];
    else if (col < K1PAD)     val = 0.0f;
    else                      val = x_prev[(size_t)row * NV2 + (col - K1PAD)];
    Apack[idx] = f2bf(val);
}

__global__ void pack_W1_kernel(const float* __restrict__ W_in,
                               const float* __restrict__ W_rec,
                               uint16_t* __restrict__ W1pack) {
    size_t idx = (size_t)blockIdx.x * blockDim.x + threadIdx.x;
    if (idx >= (size_t)NV2 * KTOT) return;
    int row = (int)(idx / KTOT);
    int col = (int)(idx % KTOT);
    float val;
    if (col < IN_DIM)         val = W_in[(size_t)row * IN_DIM + col];
    else if (col < K1PAD)     val = 0.0f;
    else                      val = W_rec[(size_t)row * NV2 + (col - K1PAD)];
    W1pack[idx] = f2bf(val);
}

__global__ void pack_W2_kernel(const float* __restrict__ W_mu,
                               const float* __restrict__ W_fb,
                               const float* __restrict__ W_pi,
                               uint16_t* __restrict__ W2pack) {
    size_t idx = (size_t)blockIdx.x * blockDim.x + threadIdx.x;
    if (idx >= (size_t)N2PAD * NV2) return;
    int row = (int)(idx / NV2);
    int col = (int)(idx % NV2);
    float val;
    if (row < 180)       val = W_mu[(size_t)row * NV2 + col];
    else if (row < 360)  val = W_fb[(size_t)(row - 180) * NV2 + col];
    else if (row == 360) val = W_pi[col];
    else                 val = 0.0f;
    W2pack[idx] = f2bf(val);
}

// ---------------------------------------------------------------------------
// Fragment assembly from LDS, per ISA 7.12.2 16-bit layouts (wave32):
//  A 16x32: lane l: row = l&15; element i -> k = i + (i>=8?8:0) + (l>>4)*8
//  B 32x16: lane l: col = l&15; element i -> k = i + (i>=8?8:0) + (l>>4)*8
// ---------------------------------------------------------------------------
static __device__ __forceinline__ v16bf load_afrag(const uint16_t* tile,
                                                   int rowOff, int lane) {
    int r = lane & 15, hf = lane >> 4;
    const uint16_t* p = tile + (size_t)(rowOff + r) * LDS_STRIDE;
    union { v16bf v; bf16x8 h[2]; } u;
    u.h[0] = *(const bf16x8*)(p + hf * 8);
    u.h[1] = *(const bf16x8*)(p + 16 + hf * 8);
    return u.v;
}

static __device__ __forceinline__ v16bf load_bfrag(const uint16_t* tile,
                                                   int colOff, int lane) {
    int n = lane & 15, hf = lane >> 4;
    const uint16_t* p = tile + (size_t)(colOff + n) * LDS_STRIDE;
    union { v16bf v; bf16x8 h[2]; } u;
    u.h[0] = *(const bf16x8*)(p + hf * 8);
    u.h[1] = *(const bf16x8*)(p + 16 + hf * 8);
    return u.v;
}

// ---------------------------------------------------------------------------
// GEMM1 + fused spiking update.
// Block tile 128x128, 256 threads = 8 wave32 waves in a 4x2 grid,
// each wave owns a 32x64 sub-tile = 2x4 WMMA accumulators.
// Tiles double-buffered in LDS, filled by async b128 loads (2 per matrix
// per thread per K-step), one barrier per K-step.
// ---------------------------------------------------------------------------
__global__ void __launch_bounds__(256)
gemm1_spike_kernel(const uint16_t* __restrict__ Apack,
                   const uint16_t* __restrict__ W1pack,
                   const float* __restrict__ b_in,
                   const float* __restrict__ v_prev,
                   const float* __restrict__ z_prev,
                   const float* __restrict__ x_prev,
                   const float* __restrict__ b_prev,
                   const float* __restrict__ adapt_mask,
                   float* __restrict__ out,
                   uint16_t* __restrict__ xbf) {
    __shared__ uint16_t As[2][128 * LDS_STRIDE];
    __shared__ uint16_t Ws[2][128 * LDS_STRIDE];

    const int tid  = threadIdx.x;
    const int lane = tid & 31;
    const int wave = tid >> 5;
    const int wr   = wave & 3;          // wave row (0..3) -> 32 rows each
    const int wc   = wave >> 2;         // wave col (0..1) -> 64 cols each
    const int blockM = blockIdx.y * 128;
    const int blockN = blockIdx.x * 128;

    // Async-copy assignment: tile = 128 rows x 4 b128-units; thread covers
    // unit (r0,q0) and (r0+64,q0).
    const int r0 = tid >> 2;            // 0..63
    const int q0 = tid & 3;             // 0..3 (16 B units within a row)
    const uint32_t offL0 = (uint32_t)(r0 * (LDS_STRIDE * 2) + q0 * 16);
    const uint32_t offL1 = offL0 + 64u * (LDS_STRIDE * 2);
    const uint32_t ldsA0 = lds_addr_of(&As[0][0]);
    const uint32_t ldsA1 = lds_addr_of(&As[1][0]);
    const uint32_t ldsW0 = lds_addr_of(&Ws[0][0]);
    const uint32_t ldsW1 = lds_addr_of(&Ws[1][0]);

    const uint64_t rowBytes = (uint64_t)KTOT * 2;           // 9280
    const uint64_t gA0 = (uint64_t)(uintptr_t)Apack
                       + (uint64_t)(blockM + r0) * rowBytes + (uint64_t)q0 * 16;
    const uint64_t gW0 = (uint64_t)(uintptr_t)W1pack
                       + (uint64_t)(blockN + r0) * rowBytes + (uint64_t)q0 * 16;

    v8f acc[2][4];
#pragma unroll
    for (int i = 0; i < 2; ++i)
#pragma unroll
        for (int j = 0; j < 4; ++j) acc[i][j] = v8f{};

    // Prologue: fill buffer 0 with K-step 0.
    async_ld_b128(ldsA0 + offL0, gA0);
    async_ld_b128(ldsA0 + offL1, gA0 + 64 * rowBytes);
    async_ld_b128(ldsW0 + offL0, gW0);
    async_ld_b128(ldsW0 + offL1, gW0 + 64 * rowBytes);
    wait_async0();
    __syncthreads();

    for (int kt = 0; kt < KSTEPS1; ++kt) {
        const int cur = kt & 1;
        if (kt + 1 < KSTEPS1) {         // prefetch next K-step into other buf
            const uint64_t ko = (uint64_t)(kt + 1) * 64;    // 32 bf16 = 64 B
            const uint32_t la = cur ? ldsA0 : ldsA1;
            const uint32_t lw = cur ? ldsW0 : ldsW1;
            async_ld_b128(la + offL0, gA0 + ko);
            async_ld_b128(la + offL1, gA0 + 64 * rowBytes + ko);
            async_ld_b128(lw + offL0, gW0 + ko);
            async_ld_b128(lw + offL1, gW0 + 64 * rowBytes + ko);
        }

        const uint16_t* At = As[cur];
        const uint16_t* Wt = Ws[cur];
        v16bf a0 = load_afrag(At, wr * 32, lane);
        v16bf a1 = load_afrag(At, wr * 32 + 16, lane);
#pragma unroll
        for (int tj = 0; tj < 4; ++tj) {
            v16bf b = load_bfrag(Wt, wc * 64 + tj * 16, lane);
            acc[0][tj] = __builtin_amdgcn_wmma_f32_16x16x32_bf16(
                false, a0, false, b, (short)0, acc[0][tj], false, false);
            acc[1][tj] = __builtin_amdgcn_wmma_f32_16x16x32_bf16(
                false, a1, false, b, (short)0, acc[1][tj], false, false);
        }

        wait_async0();                  // my prefetch landed
        __syncthreads();                // everyone's prefetch landed
    }

    // Fused epilogue: spiking neuron update, fp32.
#pragma unroll
    for (int ti = 0; ti < 2; ++ti)
#pragma unroll
        for (int tj = 0; tj < 4; ++tj)
#pragma unroll
            for (int e = 0; e < 8; ++e) {
                int m = wr * 32 + ti * 16 + (lane >> 4) * 8 + e;
                int n = wc * 64 + tj * 16 + (lane & 15);
                int gm = blockM + m;               // batch row
                int gn = blockN + n;               // neuron index
                size_t idx = (size_t)gm * NV2 + gn;
                float drive = acc[ti][tj][e] + b_in[gn];
                float vp = v_prev[idx], zp = z_prev[idx];
                float xp = x_prev[idx], bp = b_prev[idx];
                float v = BETA_MEM * vp + drive - zp;           // V_THRESH=1
                float Bth = 1.0f + BETA_AD * bp;
                float z = (v - Bth > 0.0f) ? 1.0f : 0.0f;
                float b = (RHO_AD * bp + ONE_M_RHO * z) * adapt_mask[gn];
                float x = ALPHA_F * xp + z;
                out[OFF_V + idx] = v;
                out[OFF_Z + idx] = z;
                out[OFF_X + idx] = x;
                out[OFF_B + idx] = b;
                xbf[idx] = f2bf(x);
            }
}

// ---------------------------------------------------------------------------
// GEMM2: y[4096 x 384] = x_bf16 @ W2.T + bias2 (fp32 out to workspace).
// Same skeleton as GEMM1 (async double-buffered tiles).
// ---------------------------------------------------------------------------
__global__ void __launch_bounds__(256)
gemm2_kernel(const uint16_t* __restrict__ xbf,
             const uint16_t* __restrict__ W2pack,
             const float* __restrict__ b_mu,
             const float* __restrict__ b_fb,
             const float* __restrict__ b_pi,
             float* __restrict__ y) {
    __shared__ uint16_t As[2][128 * LDS_STRIDE];
    __shared__ uint16_t Ws[2][128 * LDS_STRIDE];

    const int tid  = threadIdx.x;
    const int lane = tid & 31;
    const int wave = tid >> 5;
    const int wr   = wave & 3;
    const int wc   = wave >> 2;
    const int blockM = blockIdx.y * 128;
    const int blockN = blockIdx.x * 128;

    const int r0 = tid >> 2;
    const int q0 = tid & 3;
    const uint32_t offL0 = (uint32_t)(r0 * (LDS_STRIDE * 2) + q0 * 16);
    const uint32_t offL1 = offL0 + 64u * (LDS_STRIDE * 2);
    const uint32_t ldsA0 = lds_addr_of(&As[0][0]);
    const uint32_t ldsA1 = lds_addr_of(&As[1][0]);
    const uint32_t ldsW0 = lds_addr_of(&Ws[0][0]);
    const uint32_t ldsW1 = lds_addr_of(&Ws[1][0]);

    const uint64_t rowBytes = (uint64_t)NV2 * 2;            // 8192
    const uint64_t gA0 = (uint64_t)(uintptr_t)xbf
                       + (uint64_t)(blockM + r0) * rowBytes + (uint64_t)q0 * 16;
    const uint64_t gW0 = (uint64_t)(uintptr_t)W2pack
                       + (uint64_t)(blockN + r0) * rowBytes + (uint64_t)q0 * 16;

    v8f acc[2][4];
#pragma unroll
    for (int i = 0; i < 2; ++i)
#pragma unroll
        for (int j = 0; j < 4; ++j) acc[i][j] = v8f{};

    async_ld_b128(ldsA0 + offL0, gA0);
    async_ld_b128(ldsA0 + offL1, gA0 + 64 * rowBytes);
    async_ld_b128(ldsW0 + offL0, gW0);
    async_ld_b128(ldsW0 + offL1, gW0 + 64 * rowBytes);
    wait_async0();
    __syncthreads();

    for (int kt = 0; kt < KSTEPS2; ++kt) {
        const int cur = kt & 1;
        if (kt + 1 < KSTEPS2) {
            const uint64_t ko = (uint64_t)(kt + 1) * 64;
            const uint32_t la = cur ? ldsA0 : ldsA1;
            const uint32_t lw = cur ? ldsW0 : ldsW1;
            async_ld_b128(la + offL0, gA0 + ko);
            async_ld_b128(la + offL1, gA0 + 64 * rowBytes + ko);
            async_ld_b128(lw + offL0, gW0 + ko);
            async_ld_b128(lw + offL1, gW0 + 64 * rowBytes + ko);
        }

        const uint16_t* At = As[cur];
        const uint16_t* Wt = Ws[cur];
        v16bf a0 = load_afrag(At, wr * 32, lane);
        v16bf a1 = load_afrag(At, wr * 32 + 16, lane);
#pragma unroll
        for (int tj = 0; tj < 4; ++tj) {
            v16bf b = load_bfrag(Wt, wc * 64 + tj * 16, lane);
            acc[0][tj] = __builtin_amdgcn_wmma_f32_16x16x32_bf16(
                false, a0, false, b, (short)0, acc[0][tj], false, false);
            acc[1][tj] = __builtin_amdgcn_wmma_f32_16x16x32_bf16(
                false, a1, false, b, (short)0, acc[1][tj], false, false);
        }

        wait_async0();
        __syncthreads();
    }

#pragma unroll
    for (int ti = 0; ti < 2; ++ti)
#pragma unroll
        for (int tj = 0; tj < 4; ++tj)
#pragma unroll
            for (int e = 0; e < 8; ++e) {
                int gm = blockM + wr * 32 + ti * 16 + (lane >> 4) * 8 + e;
                int gn = blockN + wc * 64 + tj * 16 + (lane & 15);
                float bias = 0.0f;
                if (gn < 180)       bias = b_mu[gn];
                else if (gn < 360)  bias = b_fb[gn - 180];
                else if (gn == 360) bias = b_pi[0];
                y[(size_t)gm * N2PAD + gn] = acc[ti][tj][e] + bias;
            }
}

// ---------------------------------------------------------------------------
// Finalize: per batch row -> softmax(mu), copy fb, clamped softplus(pi).
// ---------------------------------------------------------------------------
__global__ void __launch_bounds__(256)
finalize_kernel(const float* __restrict__ y, float* __restrict__ out) {
    __shared__ float sm[256];
    const int row = blockIdx.x;
    const int tid = threadIdx.x;
    const float* yr = y + (size_t)row * N2PAD;

    float v = (tid < NN) ? yr[tid] : -3.402823466e38f;
    sm[tid] = v;
    __syncthreads();
#pragma unroll
    for (int s = 128; s > 0; s >>= 1) {
        if (tid < s) sm[tid] = fmaxf(sm[tid], sm[tid + s]);
        __syncthreads();
    }
    float mx = sm[0];
    __syncthreads();

    float e = (tid < NN) ? expf(yr[tid] - mx) : 0.0f;
    sm[tid] = e;
    __syncthreads();
#pragma unroll
    for (int s = 128; s > 0; s >>= 1) {
        if (tid < s) sm[tid] = sm[tid] + sm[tid + s];
        __syncthreads();
    }
    float denom = sm[0];

    if (tid < NN) {
        out[OFF_MU + (size_t)row * NN + tid] = e / denom;
        out[OFF_FB + (size_t)row * NN + tid] = yr[NN + tid];
    }
    if (tid == 0) {
        float t = yr[360];
        float sp = fmaxf(t, 0.0f) + log1pf(expf(-fabsf(t)));  // stable softplus
        out[OFF_PI + row] = fminf(sp, PI_MAX);
    }
}

// ---------------------------------------------------------------------------
extern "C" void kernel_launch(void* const* d_in, const int* in_sizes, int n_in,
                              void* d_out, int out_size, void* d_ws, size_t ws_size,
                              hipStream_t stream) {
    (void)in_sizes; (void)n_in; (void)out_size; (void)ws_size;
    const float* x_l4       = (const float*)d_in[0];
    const float* x_l23      = (const float*)d_in[1];
    const float* cue        = (const float*)d_in[2];
    const float* task_state = (const float*)d_in[3];
    const float* v_prev     = (const float*)d_in[4];
    const float* z_prev     = (const float*)d_in[5];
    const float* x_prev     = (const float*)d_in[6];
    const float* b_prev     = (const float*)d_in[7];
    const float* adapt_mask = (const float*)d_in[8];
    const float* W_in       = (const float*)d_in[9];
    const float* b_in       = (const float*)d_in[10];
    const float* W_rec      = (const float*)d_in[11];
    const float* W_mu       = (const float*)d_in[12];
    const float* b_mu       = (const float*)d_in[13];
    const float* W_fb       = (const float*)d_in[14];
    const float* b_fb       = (const float*)d_in[15];
    const float* W_pi       = (const float*)d_in[16];
    const float* b_pi       = (const float*)d_in[17];
    float* out = (float*)d_out;

    // Workspace carve-up (bf16 stores as uint16_t), ~114 MB total.
    uint16_t* Apack  = (uint16_t*)d_ws;
    uint16_t* W1pack = Apack  + (size_t)B_BATCH * KTOT;
    uint16_t* xbf    = W1pack + (size_t)NV2 * KTOT;
    uint16_t* W2pack = xbf    + (size_t)B_BATCH * NV2;
    float*    y      = (float*)(W2pack + (size_t)N2PAD * NV2);

    {
        size_t total = (size_t)B_BATCH * KTOT;
        pack_A_kernel<<<dim3((unsigned)((total + 255) / 256)), dim3(256), 0, stream>>>(
            x_l4, x_l23, cue, task_state, x_prev, Apack);
    }
    {
        size_t total = (size_t)NV2 * KTOT;
        pack_W1_kernel<<<dim3((unsigned)((total + 255) / 256)), dim3(256), 0, stream>>>(
            W_in, W_rec, W1pack);
    }
    {
        size_t total = (size_t)N2PAD * NV2;
        pack_W2_kernel<<<dim3((unsigned)((total + 255) / 256)), dim3(256), 0, stream>>>(
            W_mu, W_fb, W_pi, W2pack);
    }

    gemm1_spike_kernel<<<dim3(NV2 / 128, B_BATCH / 128), dim3(256), 0, stream>>>(
        Apack, W1pack, b_in, v_prev, z_prev, x_prev, b_prev, adapt_mask, out, xbf);

    gemm2_kernel<<<dim3(N2PAD / 128, B_BATCH / 128), dim3(256), 0, stream>>>(
        xbf, W2pack, b_mu, b_fb, b_pi, y);

    finalize_kernel<<<dim3(B_BATCH), dim3(256), 0, stream>>>(y, out);
}
